// AttentionEinOps_40089224741074
// MI455X (gfx1250) — matrix operand order repaired and checked by reading, not compiled, not run on hardware
//
#include <hip/hip_runtime.h>
#include <hip/hip_bf16.h>

// ---------------------------------------------------------------------------
// Problem constants (match reference)
// ---------------------------------------------------------------------------
#define BB      2
#define SQ      2048
#define SKK     2048
#define DMODEL  1024
#define NHEADS  16
#define DHEAD   64
#define LN_EPS  1e-5f
#define NEG_BIG (-1e30f)

typedef __attribute__((ext_vector_type(16))) __bf16 v16bf;
typedef __attribute__((ext_vector_type(8)))  __bf16 v8bf;
typedef __attribute__((ext_vector_type(8)))  float  v8f;

#define WMMA_BF16(A, B, C) \
  __builtin_amdgcn_wmma_f32_16x16x32_bf16(false, (A), false, (B), (short)0, (C), false, false)

#define CAT16(lo, hi) \
  __builtin_shufflevector((lo), (hi), 0, 1, 2, 3, 4, 5, 6, 7, 8, 9, 10, 11, 12, 13, 14, 15)

// ---------------------------------------------------------------------------
// Fragment loaders (CDNA5 ISA 7.12.2, wave32). Both are contiguous b128 pairs.
// ---------------------------------------------------------------------------
// A 16x32 bf16 from a row-major [M][K] tile. Per lane: K = half*16+{0..15}
// -> one 32-byte run (matches load_BT_frag).
__device__ __forceinline__ v16bf load_A_frag(const __bf16* lds, int ld,
                                             int r0, int k0, int lane) {
  const int M = lane & 15, half = lane >> 4;
  const __bf16* base = lds + (r0 + M) * ld + k0 + half * 16;
  const v8bf lo = *(const v8bf*)base;
  const v8bf hi = *(const v8bf*)(base + 8);
  return CAT16(lo, hi);
}

// B 32x16 bf16 from an N-major [N][K] tile. Per lane (col n = lane&15):
// K = half*16+{0..15} -> one 32-byte run.
__device__ __forceinline__ v16bf load_BT_frag(const __bf16* lds, int ld,
                                              int n0, int k0, int lane) {
  const int n = lane & 15, half = lane >> 4;
  const __bf16* base = lds + (n0 + n) * ld + k0 + half * 16;
  const v8bf lo = *(const v8bf*)base;
  const v8bf hi = *(const v8bf*)(base + 8);
  return CAT16(lo, hi);
}

// Async global -> LDS b128 copy (CDNA5 GLOBAL_LOAD_ASYNC_TO_LDS_B128, GVS
// addressing: uniform 64-bit SGPR base + per-lane 32-bit byte offset; the
// LDS destination byte offset rides in the VDST VGPR). Tracked by ASYNCcnt.
__device__ __forceinline__ void async_copy_b128(unsigned lds_off,
                                                const void* sbase,
                                                unsigned goff) {
  asm volatile("global_load_async_to_lds_b128 %0, %1, %2"
               :
               : "v"(lds_off), "v"(goff), "s"(sbase)
               : "memory");
}

__device__ __forceinline__ void async_wait0() {
  asm volatile("s_wait_asynccnt 0" ::: "memory");
}

// ---------------------------------------------------------------------------
// Kernel 1: QKV projection + bias + per-head LayerNorm (Q,K) per 64x64 tile.
// grid = (B*SQ/64, NHEADS, 3)  z: 0=Q 1=K 2=V.  block = 256 (8 waves).
// ---------------------------------------------------------------------------
__global__ __launch_bounds__(256) void qkv_ln_kernel(
    const float* __restrict__ x_q, const float* __restrict__ x_kv,
    const float* __restrict__ W_Q, const float* __restrict__ W_K,
    const float* __restrict__ W_V, const float* __restrict__ b_Q,
    const float* __restrict__ b_K, const float* __restrict__ b_V,
    const float* __restrict__ ln1w, const float* __restrict__ ln1b,
    const float* __restrict__ ln2w, const float* __restrict__ ln2b,
    float* __restrict__ k_out, float* __restrict__ v_out,
    __bf16* __restrict__ q_bf, __bf16* __restrict__ k_bf,
    __bf16* __restrict__ v_t) {
  const int z    = blockIdx.z;
  const int head = blockIdx.y;
  const int tile = blockIdx.x;
  const int t    = threadIdx.x;
  const int lane = t & 31, wave = t >> 5;
  const int rt = wave >> 1, c0 = (wave & 1) * 32;
  const int half = lane >> 4, nn = lane & 15;
  const int bidx = (tile * 64) / SQ;
  const int s0   = (tile * 64) % SQ;

  const float* X    = (z == 0) ? x_q : x_kv;
  const float* W    = (z == 0) ? W_Q : (z == 1 ? W_K : W_V);
  const float* bias = (z == 0) ? b_Q : (z == 1 ? b_K : b_V);
  const float* lw   = (z == 0) ? ln1w : ln2w;
  const float* lb   = (z == 0) ? ln1b : ln2b;

  __shared__ __bf16 lA[64][40];    // 64 tokens x 32 K, row-major (A operand)
  __shared__ __bf16 lBt[64][40];   // 64 head-cols x 32 K, N-major (B operand)
  __shared__ float  lT[64][65];    // f32 output tile for LN
  __shared__ float  lMu[64], lRs[64];

  v8f acc0 = {}, acc1 = {};

  // A tile: 4 thr/row, 8 contiguous floats each (coalesced, b128-friendly)
  const int arow = t >> 2, ac0 = (t & 3) * 8;
  // B tile transposed-from-global: thread owns one output col, 8 consecutive K
  const int bcol = t & 63, bkr = (t >> 6) * 8;
  const float* xrow = X + ((size_t)bidx * SQ + s0 + arow) * DMODEL + ac0;
  const float* wcol = W + (size_t)head * DMODEL * DHEAD + bkr * DHEAD + bcol;

  for (int kk = 0; kk < DMODEL; kk += 32) {
#pragma unroll
    for (int i = 0; i < 8; ++i) lA[arow][ac0 + i] = (__bf16)xrow[kk + i];
#pragma unroll
    for (int i = 0; i < 8; ++i)
      lBt[bcol][bkr + i] = (__bf16)wcol[(size_t)(kk + i) * DHEAD];
    __syncthreads();

    const v16bf a  = load_A_frag(&lA[0][0], 40, rt * 16, 0, lane);
    const v16bf b0 = load_BT_frag(&lBt[0][0], 40, c0, 0, lane);
    const v16bf b1 = load_BT_frag(&lBt[0][0], 40, c0 + 16, 0, lane);
    acc0 = WMMA_BF16(a, b0, acc0);
    acc1 = WMMA_BF16(a, b1, acc1);
    __syncthreads();
  }

  // bias + park tile in LDS (C layout: M = v + 8*(lane>>4))
  const float bv0 = bias[head * DHEAD + c0 + nn];
  const float bv1 = bias[head * DHEAD + c0 + 16 + nn];
#pragma unroll
  for (int v = 0; v < 8; ++v) {
    const int r = rt * 16 + v + half * 8;
    lT[r][c0 + nn]      = acc0[v] + bv0;
    lT[r][c0 + 16 + nn] = acc1[v] + bv1;
  }
  __syncthreads();

  // per-head LayerNorm stats (one thread per token row)
  if (t < 64) {
    if (z < 2) {
      float s = 0.f;
      for (int c = 0; c < DHEAD; ++c) s += lT[t][c];
      const float mu = s * (1.0f / DHEAD);
      float var = 0.f;
      for (int c = 0; c < DHEAD; ++c) {
        const float d = lT[t][c] - mu;
        var += d * d;
      }
      lMu[t] = mu;
      lRs[t] = rsqrtf(var * (1.0f / DHEAD) + LN_EPS);
    } else {
      lMu[t] = 0.f;
      lRs[t] = 1.f;
    }
  }
  __syncthreads();

  const int orow = t >> 2, ocb = (t & 3) * 16;
  const float mu = lMu[orow], rs = lRs[orow];
  const int s = s0 + orow;
#pragma unroll
  for (int i = 0; i < 16; ++i) {
    const int c = ocb + i;
    float val = lT[orow][c];
    if (z < 2) val = (val - mu) * rs * lw[c] + lb[c];
    if (z == 0) {
      q_bf[(((size_t)bidx * NHEADS + head) * SQ + s) * DHEAD + c] = (__bf16)val;
    } else if (z == 1) {
      k_out[(((size_t)bidx * SKK + s) * NHEADS + head) * DHEAD + c] = val;
      k_bf[(((size_t)bidx * NHEADS + head) * SKK + s) * DHEAD + c] = (__bf16)val;
    } else {
      v_out[(((size_t)bidx * SKK + s) * NHEADS + head) * DHEAD + c] = val;
      // pre-transposed V: [b, head, h, s] so attention's LDS tile load is
      // contiguous on both sides (PV B-operand wants N-major = [h][key])
      v_t[(((size_t)bidx * NHEADS + head) * DHEAD + c) * SKK + s] = (__bf16)val;
    }
  }
}

// ---------------------------------------------------------------------------
// Kernel 2: causal flash attention per (b, head, 64-query tile).
// grid = (SQ/64, NHEADS, B).  block = 256 (8 waves).
// K/V tiles are staged with async global->LDS copies (ASYNCcnt).
// ---------------------------------------------------------------------------
__global__ __launch_bounds__(256) void attn_kernel(
    const __bf16* __restrict__ q_bf, const __bf16* __restrict__ k_bf,
    const __bf16* __restrict__ v_t, __bf16* __restrict__ z_bf) {
  const int b    = blockIdx.z;
  const int head = blockIdx.y;
  const int q0   = blockIdx.x * 64;
  const int t    = threadIdx.x;
  const int lane = t & 31, wave = t >> 5;
  const int rt = wave >> 1, c0 = (wave & 1) * 32;
  const int half = lane >> 4, nn = lane & 15;

  __shared__ __bf16 lQ[64][72];    // [query][h]   A operand (row-major)
  __shared__ __bf16 lK[64][72];    // [key][h]     B operand for S (N-major!)
  __shared__ __bf16 lVt[64][72];   // [h][key]     B operand for PV (N-major)
  __shared__ __bf16 lP[64][72];    // [query][key] A operand (row-major)
  __shared__ float  lS[64][65];
  __shared__ float  lM[64], lL[64], lC[64];
  __shared__ float  lRmax[64][4], lRsum[64][4];

  const __bf16* qbase = q_bf + (((size_t)b * NHEADS + head) * SQ + q0) * DHEAD;
  const __bf16* kbase = k_bf + ((size_t)b * NHEADS + head) * SKK * DHEAD;
  const __bf16* vtb   = v_t + ((size_t)b * NHEADS + head) * DHEAD * SKK;

  const int row = t >> 2, cb = (t & 3) * 16;  // 4 thr/row * 16 bf16 (32 B)
#pragma unroll
  for (int i = 0; i < 16; ++i) lQ[row][cb + i] = qbase[row * DHEAD + cb + i];
  if (t < 64) { lM[t] = NEG_BIG; lL[t] = 0.f; }

  // per-lane LDS byte offsets of this thread's 32-byte slice of each tile
  const unsigned ldsK = (unsigned)(size_t)&lK[row][cb];
  const unsigned ldsV = (unsigned)(size_t)&lVt[row][cb];
  const unsigned goffK0 = (unsigned)(row * DHEAD + cb) * sizeof(__bf16);
  const unsigned goffV0 = (unsigned)(row * SKK + cb) * sizeof(__bf16);

  v8f za0 = {}, za1 = {};

  const int nsteps = blockIdx.x + 1;  // causal: key tiles 0..q-tile
  for (int step = 0; step < nsteps; ++step) {
    const int k0 = step * 64;
    __syncthreads();  // retire previous iteration's reads of lK/lVt/lP

    // async-stage K and V tiles straight into LDS (2 x b128 each per lane)
    const unsigned gk = goffK0 + (unsigned)(k0 * DHEAD) * sizeof(__bf16);
    const unsigned gv = goffV0 + (unsigned)k0 * sizeof(__bf16);
    async_copy_b128(ldsK, kbase, gk);
    async_copy_b128(ldsK + 16, kbase, gk + 16);
    async_copy_b128(ldsV, vtb, gv);
    async_copy_b128(ldsV + 16, vtb, gv + 16);
    if (step + 1 < nsteps) {  // warm L2 for the next tile
      __builtin_prefetch(kbase + (size_t)(k0 + 64 + row) * DHEAD, 0, 0);
      __builtin_prefetch(vtb + (size_t)row * SKK + k0 + 64 + cb, 0, 0);
    }
    async_wait0();
    __syncthreads();

    // S = Q * K^T : A = lQ rows, B(k=h, n=key) = lK[key][h] (N-major)
    v8f sa0 = {}, sa1 = {};
#pragma unroll
    for (int h = 0; h < DHEAD; h += 32) {
      const v16bf a  = load_A_frag(&lQ[0][0], 72, rt * 16, h, lane);
      const v16bf b0 = load_BT_frag(&lK[0][0], 72, c0, h, lane);
      const v16bf b1 = load_BT_frag(&lK[0][0], 72, c0 + 16, h, lane);
      sa0 = WMMA_BF16(a, b0, sa0);
      sa1 = WMMA_BF16(a, b1, sa1);
    }

    // causal mask + spill scores
#pragma unroll
    for (int v = 0; v < 8; ++v) {
      const int r  = rt * 16 + v + half * 8;
      const int qg = q0 + r;
      lS[r][c0 + nn]      = (qg >= k0 + c0 + nn)      ? sa0[v] : NEG_BIG;
      lS[r][c0 + 16 + nn] = (qg >= k0 + c0 + 16 + nn) ? sa1[v] : NEG_BIG;
    }
    __syncthreads();

    // online softmax, 4 threads per query row (two-stage LDS reduction)
    {
      const int rr = t >> 2, qd = t & 3;
      float pm = NEG_BIG;
#pragma unroll
      for (int i = 0; i < 16; ++i) pm = fmaxf(pm, lS[rr][qd * 16 + i]);
      lRmax[rr][qd] = pm;
      __syncthreads();
      const float mn =
          fmaxf(lM[rr], fmaxf(fmaxf(lRmax[rr][0], lRmax[rr][1]),
                              fmaxf(lRmax[rr][2], lRmax[rr][3])));
      float ps = 0.f;
#pragma unroll
      for (int i = 0; i < 16; ++i) {
        const float p = __expf(lS[rr][qd * 16 + i] - mn);
        lP[rr][qd * 16 + i] = (__bf16)p;
        ps += p;
      }
      lRsum[rr][qd] = ps;
      __syncthreads();
      if (qd == 0) {
        const float corr = __expf(lM[rr] - mn);
        lL[rr] = lL[rr] * corr + lRsum[rr][0] + lRsum[rr][1] + lRsum[rr][2] +
                 lRsum[rr][3];
        lM[rr] = mn;
        lC[rr] = corr;
      }
    }
    __syncthreads();

    // rescale running output, then Z += P * V
#pragma unroll
    for (int v = 0; v < 8; ++v) {
      const float f = lC[rt * 16 + v + half * 8];
      za0[v] *= f;
      za1[v] *= f;
    }
#pragma unroll
    for (int kc = 0; kc < 64; kc += 32) {
      const v16bf a  = load_A_frag(&lP[0][0], 72, rt * 16, kc, lane);
      const v16bf b0 = load_BT_frag(&lVt[0][0], 72, c0, kc, lane);
      const v16bf b1 = load_BT_frag(&lVt[0][0], 72, c0 + 16, kc, lane);
      za0 = WMMA_BF16(a, b0, za0);
      za1 = WMMA_BF16(a, b1, za1);
    }
  }

  // normalize and emit z in [b, s, n*h] layout (token-major for O-proj)
#pragma unroll
  for (int v = 0; v < 8; ++v) {
    const int r = rt * 16 + v + half * 8;
    const float inv = 1.0f / lL[r];
    const size_t base = ((size_t)b * SQ + q0 + r) * DMODEL + head * DHEAD;
    z_bf[base + c0 + nn]      = (__bf16)(za0[v] * inv);
    z_bf[base + c0 + 16 + nn] = (__bf16)(za1[v] * inv);
  }
}

// ---------------------------------------------------------------------------
// Kernel 3: output projection  out = Z[4096,1024] x W_O[1024,1024] + b_O.
// grid = (B*SQ/64, DMODEL/64).  block = 256.
// ---------------------------------------------------------------------------
__global__ __launch_bounds__(256) void oproj_kernel(
    const __bf16* __restrict__ z_bf, const float* __restrict__ W_O,
    const float* __restrict__ b_O, float* __restrict__ out) {
  const int tile = blockIdx.x;
  const int d0   = blockIdx.y * 64;
  const int t    = threadIdx.x;
  const int lane = t & 31, wave = t >> 5;
  const int rt = wave >> 1, c0 = (wave & 1) * 32;
  const int half = lane >> 4, nn = lane & 15;

  __shared__ __bf16 lA[64][40];    // 64 tokens x 32 K (row-major)
  __shared__ __bf16 lBt[64][40];   // 64 out-cols x 32 K (N-major)

  v8f acc0 = {}, acc1 = {};

  const int arow = t >> 2, ac0 = (t & 3) * 8;
  const int bcol = t & 63, bkr = (t >> 6) * 8;
  const __bf16* apt = z_bf + ((size_t)tile * 64 + arow) * DMODEL + ac0;
  const float*  bpt = W_O + (size_t)bkr * DMODEL + d0 + bcol;

  for (int kk = 0; kk < DMODEL; kk += 32) {
#pragma unroll
    for (int i = 0; i < 8; ++i) lA[arow][ac0 + i] = apt[kk + i];
#pragma unroll
    for (int i = 0; i < 8; ++i)
      lBt[bcol][bkr + i] = (__bf16)bpt[(size_t)(kk + i) * DMODEL];
    __syncthreads();

    const v16bf a  = load_A_frag(&lA[0][0], 40, rt * 16, 0, lane);
    const v16bf b0 = load_BT_frag(&lBt[0][0], 40, c0, 0, lane);
    const v16bf b1 = load_BT_frag(&lBt[0][0], 40, c0 + 16, 0, lane);
    acc0 = WMMA_BF16(a, b0, acc0);
    acc1 = WMMA_BF16(a, b1, acc1);
    __syncthreads();
  }

#pragma unroll
  for (int v = 0; v < 8; ++v) {
    const int r = rt * 16 + v + half * 8;
    const size_t rowoff = ((size_t)tile * 64 + r) * DMODEL;
    out[rowoff + d0 + c0 + nn]      = acc0[v] + b_O[d0 + c0 + nn];
    out[rowoff + d0 + c0 + 16 + nn] = acc1[v] + b_O[d0 + c0 + 16 + nn];
  }
}

// ---------------------------------------------------------------------------
// Host-side launcher
// ---------------------------------------------------------------------------
extern "C" void kernel_launch(void* const* d_in, const int* in_sizes, int n_in,
                              void* d_out, int out_size, void* d_ws,
                              size_t ws_size, hipStream_t stream) {
  (void)in_sizes; (void)n_in; (void)out_size; (void)ws_size;

  const float* x_q  = (const float*)d_in[0];
  const float* x_kv = (const float*)d_in[1];
  // d_in[2] is the causal mask: tril by construction, applied analytically.
  const float* W_Q  = (const float*)d_in[3];
  const float* W_K  = (const float*)d_in[4];
  const float* W_V  = (const float*)d_in[5];
  const float* W_O  = (const float*)d_in[6];
  const float* b_Q  = (const float*)d_in[7];
  const float* b_K  = (const float*)d_in[8];
  const float* b_V  = (const float*)d_in[9];
  const float* b_O  = (const float*)d_in[10];
  const float* ln1w = (const float*)d_in[11];
  const float* ln1b = (const float*)d_in[12];
  const float* ln2w = (const float*)d_in[13];
  const float* ln2b = (const float*)d_in[14];

  // d_out = (out, k, v) concatenated flat
  float* out   = (float*)d_out;
  float* k_out = out + (size_t)BB * SQ * DMODEL;
  float* v_out = out + 2 * (size_t)BB * SQ * DMODEL;

  // workspace (32 MB): q/k head-major bf16, V pre-transposed, z token-major
  const size_t NTOK = (size_t)BB * SQ * DMODEL;  // 4,194,304 elements
  __bf16* q_bf = (__bf16*)d_ws;
  __bf16* k_bf = q_bf + NTOK;
  __bf16* v_t  = q_bf + 2 * NTOK;
  __bf16* z_bf = q_bf + 3 * NTOK;

  qkv_ln_kernel<<<dim3(BB * SQ / 64, NHEADS, 3), 256, 0, stream>>>(
      x_q, x_kv, W_Q, W_K, W_V, b_Q, b_K, b_V, ln1w, ln1b, ln2w, ln2b,
      k_out, v_out, q_bf, k_bf, v_t);

  attn_kernel<<<dim3(SQ / 64, NHEADS, BB), 256, 0, stream>>>(q_bf, k_bf, v_t,
                                                             z_bf);

  oproj_kernel<<<dim3(BB * SQ / 64, DMODEL / 64), 256, 0, stream>>>(z_bf, W_O,
                                                                    b_O, out);
}
